// NALUi2_22677427323455
// MI455X (gfx1250) — matrix operand order repaired and checked by reading, not compiled, not run on hardware
//
#include <hip/hip_runtime.h>
#include <cmath>

// ---------------- types for CDNA5 WMMA ----------------
typedef __bf16 bf16_t;
typedef __attribute__((ext_vector_type(16))) __bf16 v16bf;
typedef __attribute__((ext_vector_type(8)))  __bf16 v8bf;
typedef __attribute__((ext_vector_type(8)))  float  v8f;

union Frag { v16bf v; v8bf h[2]; };

#define WMMA_BF16(acc, A, B) \
    acc = __builtin_amdgcn_wmma_f32_16x16x32_bf16(false, (A), false, (B), (short)0, (acc), false, false)

// ---------------- problem constants ----------------
constexpr int IN  = 1024;
constexpr int OUT = 1024;
constexpr int BATCH = 256;

constexpr size_t AE = (size_t)BATCH * IN;   // 262144 elems (A-side matrices)
constexpr size_t BE = (size_t)IN * OUT;     // 1048576 elems (B-side matrices)

// workspace element offsets (bf16 elements)
constexpr size_t XH  = 0;           // bf16 hi of x            (256 x 1024)
constexpr size_t XL  = 1 * AE;      // bf16 lo of x
constexpr size_t LXH = 2 * AE;      // hi of log(max(|x|,1e-7))
constexpr size_t LXL = 3 * AE;      // lo of same
constexpr size_t NG  = 4 * AE;      // 1.0 if x<0 else 0.0
constexpr size_t B1H = 5 * AE;            // W1^T hi  [n][k]  (1024 x 1024)
constexpr size_t B1L = 5 * AE + 1 * BE;   // W1^T lo
constexpr size_t B2H = 5 * AE + 2 * BE;   // W2^T hi
constexpr size_t B2L = 5 * AE + 3 * BE;   // W2^T lo
constexpr size_t LH  = 5 * AE + 4 * BE;   // log|1-2|W2[n][k]|| hi (natural orient)
constexpr size_t LL  = 5 * AE + 5 * BE;   // lo
constexpr size_t SS  = 5 * AE + 6 * BE;   // 1.0 if (1-2|W2[n][k]|)<0 else 0.0

// second n-tile of a wave lives 16 B^T rows further down: constant elem offset
constexpr int NT2 = 16 * IN;   // 16384 elems = 32 KB (folds into load ioffset)

__device__ __forceinline__ float sigf(float x) { return 1.0f / (1.0f + expf(-x)); }

__device__ __forceinline__ void split2(float v, bf16_t* hi, bf16_t* lo) {
    bf16_t h = (bf16_t)v;
    *hi = h;
    *lo = (bf16_t)(v - (float)h);
}

// ---------------- prep kernel: A-side operands from x ----------------
__global__ __launch_bounds__(256) void prep_a(const float* __restrict__ x,
                                              bf16_t* __restrict__ ws) {
    size_t i = (size_t)blockIdx.x * 256 + threadIdx.x;
    if (i >= AE) return;
    float v = x[i];
    bf16_t h, l;
    split2(v, &h, &l);
    ws[XH + i] = h; ws[XL + i] = l;
    float lx = logf(fmaxf(fabsf(v), 1e-7f));
    split2(lx, &h, &l);
    ws[LXH + i] = h; ws[LXL + i] = l;
    ws[NG + i] = (v < 0.0f) ? (bf16_t)1.0f : (bf16_t)0.0f;
}

// ---------------- prep kernel: natural-orientation B-side (L, S from W2) ----------
// ms[b,o] = prod_i ( sign(x[b,i])*|W2[o,i]| + 1-|W2[o,i]| )
//         = (-1)^(neg@S)[b,o] * exp( neg @ L )  with L[n=o][k=i] — same
// orientation as w_hat2 storage (row o, col i), so reads are natural.
__global__ __launch_bounds__(256) void prep_b_nat(const float* __restrict__ w2,
                                                  const float* __restrict__ m2,
                                                  bf16_t* __restrict__ ws) {
    size_t i = (size_t)blockIdx.x * 256 + threadIdx.x;
    if (i >= BE) return;
    float wv = tanhf(w2[i]) * sigf(m2[i]);
    float t  = 1.0f - 2.0f * fabsf(wv);
    ws[SS + i] = (t < 0.0f) ? (bf16_t)1.0f : (bf16_t)0.0f;
    float L = logf(fmaxf(fabsf(t), 1e-30f));   // clamp: avoid -inf (0*inf -> NaN in WMMA)
    bf16_t h, l;
    split2(L, &h, &l);
    ws[LH + i] = h; ws[LL + i] = l;
}

// ---------------- prep kernel: transposed B-side (W1^T, W2^T hi/lo) ------------
// 32x32 LDS tile transpose so GEMM B-fragments are contiguous along K.
__global__ __launch_bounds__(1024) void prep_b_tr(const float* __restrict__ w1,
                                                  const float* __restrict__ m1,
                                                  const float* __restrict__ w2,
                                                  const float* __restrict__ m2,
                                                  bf16_t* __restrict__ ws) {
    __shared__ float t1[32][33];
    __shared__ float t2[32][33];
    int tx = threadIdx.x, ty = threadIdx.y;
    int K0 = blockIdx.x * 32, N0 = blockIdx.y * 32;
    size_t iidx = (size_t)(K0 + ty) * OUT + (N0 + tx);      // coalesced read
    t1[ty][tx] = tanhf(w1[iidx]) * sigf(m1[iidx]);
    t2[ty][tx] = tanhf(w2[iidx]) * sigf(m2[iidx]);
    __syncthreads();
    size_t oidx = (size_t)(N0 + ty) * IN + (K0 + tx);       // coalesced write
    bf16_t h, l;
    split2(t1[tx][ty], &h, &l);
    ws[B1H + oidx] = h; ws[B1L + oidx] = l;
    split2(t2[tx][ty], &h, &l);
    ws[B2H + oidx] = h; ws[B2L + oidx] = l;
}

// ---------------- main fused 4-GEMM WMMA kernel ----------------
// One wave computes a 16(M) x 32(N) output strip = two 16x16 WMMA tiles,
// reusing every A fragment across both n-tiles (38 b128 loads : 18 WMMAs
// per k-step). Four f32 accumulators per tile:
//   acc_a   = x    @ W1      (bf16 hi/lo split: 3 wmma)
//   acc_lm  = logx @ W2      (3 wmma)
//   acc_lms = neg  @ L       (2 wmma; neg exact in bf16)
//   acc_par = neg  @ S       (1 wmma; exact integer parity sum)
__global__ __launch_bounds__(256) void nalu_main(const bf16_t* __restrict__ ws,
                                                 const float* __restrict__ g,
                                                 float* __restrict__ out) {
    int tid  = threadIdx.x;
    int wave = tid >> 5;
    int lane = tid & 31;
    int hf   = lane >> 4;      // lane half (ISA fragment layout)
    int lq   = lane & 15;

    int ngrp  = blockIdx.x * 8 + wave;   // 32 n-pairs total
    int mtile = blockIdx.y;              // 16 m-tiles total

    int arow = mtile * 16 + lq;          // A-matrix row (M) for this lane
    int brow = ngrp * 32 + lq;           // B^T row (output col N) tile 0; tile 1 = +16

    const bf16_t* pxh  = ws + XH  + (size_t)arow * IN;
    const bf16_t* pxl  = ws + XL  + (size_t)arow * IN;
    const bf16_t* plxh = ws + LXH + (size_t)arow * IN;
    const bf16_t* plxl = ws + LXL + (size_t)arow * IN;
    const bf16_t* png  = ws + NG  + (size_t)arow * IN;

    const bf16_t* pb1h = ws + B1H + (size_t)brow * IN;
    const bf16_t* pb1l = ws + B1L + (size_t)brow * IN;
    const bf16_t* pb2h = ws + B2H + (size_t)brow * IN;
    const bf16_t* pb2l = ws + B2L + (size_t)brow * IN;
    const bf16_t* plh  = ws + LH  + (size_t)brow * IN;
    const bf16_t* pll  = ws + LL  + (size_t)brow * IN;
    const bf16_t* pss  = ws + SS  + (size_t)brow * IN;

    v8f acc_a0 = {}, acc_lm0 = {}, acc_lms0 = {}, acc_par0 = {};
    v8f acc_a1 = {}, acc_lm1 = {}, acc_lms1 = {}, acc_par1 = {};

#pragma unroll 2
    for (int kk = 0; kk < IN / 32; ++kk) {
        // A fragment (16x32 bf16): lane holds K run [16*hf, +16)
        int ka = kk * 32 + 16 * hf;
        Frag fxh, fxl, flh, fll, fng;
        fxh.h[0] = *(const v8bf*)(pxh  + ka); fxh.h[1] = *(const v8bf*)(pxh  + ka + 8);
        fxl.h[0] = *(const v8bf*)(pxl  + ka); fxl.h[1] = *(const v8bf*)(pxl  + ka + 8);
        flh.h[0] = *(const v8bf*)(plxh + ka); flh.h[1] = *(const v8bf*)(plxh + ka + 8);
        fll.h[0] = *(const v8bf*)(plxl + ka); fll.h[1] = *(const v8bf*)(plxl + ka + 8);
        fng.h[0] = *(const v8bf*)(png  + ka); fng.h[1] = *(const v8bf*)(png  + ka + 8);

        // B fragments (32x16 bf16): lane holds K run [16*hf, +16) of B^T row n
        int kb = kk * 32 + 16 * hf;
        v16bf b1h0 = *(const v16bf*)(pb1h + kb);
        v16bf b1l0 = *(const v16bf*)(pb1l + kb);
        v16bf b2h0 = *(const v16bf*)(pb2h + kb);
        v16bf b2l0 = *(const v16bf*)(pb2l + kb);
        v16bf blh0 = *(const v16bf*)(plh  + kb);
        v16bf bll0 = *(const v16bf*)(pll  + kb);
        v16bf bss0 = *(const v16bf*)(pss  + kb);
        v16bf b1h1 = *(const v16bf*)(pb1h + NT2 + kb);
        v16bf b1l1 = *(const v16bf*)(pb1l + NT2 + kb);
        v16bf b2h1 = *(const v16bf*)(pb2h + NT2 + kb);
        v16bf b2l1 = *(const v16bf*)(pb2l + NT2 + kb);
        v16bf blh1 = *(const v16bf*)(plh  + NT2 + kb);
        v16bf bll1 = *(const v16bf*)(pll  + NT2 + kb);
        v16bf bss1 = *(const v16bf*)(pss  + NT2 + kb);

        WMMA_BF16(acc_a0,   fxh.v, b1h0);
        WMMA_BF16(acc_a1,   fxh.v, b1h1);
        WMMA_BF16(acc_a0,   fxh.v, b1l0);
        WMMA_BF16(acc_a1,   fxh.v, b1l1);
        WMMA_BF16(acc_a0,   fxl.v, b1h0);
        WMMA_BF16(acc_a1,   fxl.v, b1h1);
        WMMA_BF16(acc_lm0,  flh.v, b2h0);
        WMMA_BF16(acc_lm1,  flh.v, b2h1);
        WMMA_BF16(acc_lm0,  flh.v, b2l0);
        WMMA_BF16(acc_lm1,  flh.v, b2l1);
        WMMA_BF16(acc_lm0,  fll.v, b2h0);
        WMMA_BF16(acc_lm1,  fll.v, b2h1);
        WMMA_BF16(acc_lms0, fng.v, blh0);
        WMMA_BF16(acc_lms1, fng.v, blh1);
        WMMA_BF16(acc_lms0, fng.v, bll0);
        WMMA_BF16(acc_lms1, fng.v, bll1);
        WMMA_BF16(acc_par0, fng.v, bss0);
        WMMA_BF16(acc_par1, fng.v, bss1);
    }

    // Epilogue. C/D layout: lane lq = N col within tile; VGPR r -> M row = r + 8*hf.
    float gv0 = sigf(g[brow]);
    float gv1 = sigf(g[brow + 16]);
#pragma unroll
    for (int r = 0; r < 8; ++r) {
        int m = mtile * 16 + r + 8 * hf;
        // tile 0
        {
            float mval = expf(fminf(acc_lm0[r], 20.0f));
            float msg  = expf(acc_lms0[r]);          // |ms| (<=1: all logs <= 0)
            int   p    = (int)(acc_par0[r] + 0.5f);  // exact integer negative-count
            float ms   = (p & 1) ? -msg : msg;
            ms = fminf(fmaxf(ms, -1.0f), 1.0f);
            out[(size_t)m * OUT + brow] = gv0 * acc_a0[r] + (1.0f - gv0) * (mval * ms);
        }
        // tile 1
        {
            float mval = expf(fminf(acc_lm1[r], 20.0f));
            float msg  = expf(acc_lms1[r]);
            int   p    = (int)(acc_par1[r] + 0.5f);
            float ms   = (p & 1) ? -msg : msg;
            ms = fminf(fmaxf(ms, -1.0f), 1.0f);
            out[(size_t)m * OUT + brow + 16] = gv1 * acc_a1[r] + (1.0f - gv1) * (mval * ms);
        }
    }
}

// ---------------- host entry ----------------
extern "C" void kernel_launch(void* const* d_in, const int* in_sizes, int n_in,
                              void* d_out, int out_size, void* d_ws, size_t ws_size,
                              hipStream_t stream) {
    (void)in_sizes; (void)n_in; (void)out_size; (void)ws_size;
    const float* x  = (const float*)d_in[0];
    const float* w1 = (const float*)d_in[1];
    const float* m1 = (const float*)d_in[2];
    const float* w2 = (const float*)d_in[3];
    const float* m2 = (const float*)d_in[4];
    const float* g  = (const float*)d_in[5];
    bf16_t* ws = (bf16_t*)d_ws;
    float* out = (float*)d_out;

    prep_a    <<<dim3((unsigned)(AE / 256)), 256, 0, stream>>>(x, ws);
    prep_b_nat<<<dim3((unsigned)(BE / 256)), 256, 0, stream>>>(w2, m2, ws);
    prep_b_tr <<<dim3(IN / 32, OUT / 32), dim3(32, 32), 0, stream>>>(w1, m1, w2, m2, ws);
    // 64 n-tiles -> 32 n-pairs; 8 waves/block handle 8 n-pairs; 16 m-tiles
    nalu_main <<<dim3(4, BATCH / 16), 256, 0, stream>>>(ws, g, out);
}
